// RNN_19825569039021
// MI455X (gfx1250) — matrix operand order repaired and checked by reading, not compiled, not run on hardware
//
#include <hip/hip_runtime.h>
#include <hip/hip_bf16.h>
#include <stdint.h>

// Problem constants (match reference)
#define VOCAB 33
#define VPAD  48        // VOCAB padded to 3 WMMA N-tiles
#define EMB   256
#define HID   512
#define GATES 1536      // 3*HID
#define BATCH 2048
#define SEQT  140
#define ROWS  16        // batch rows per block (one WMMA M tile)
#define NTILES 96       // GATES / 16

typedef __attribute__((ext_vector_type(16))) __bf16   v16bf;
typedef __attribute__((ext_vector_type(8)))  float    v8f;
typedef uint32_t v4u __attribute__((ext_vector_type(4)));

union ABPack { v16bf v; v4u q[2]; };

#define WMMA_BF16(Adata, Bdata, Cacc) \
    __builtin_amdgcn_wmma_f32_16x16x32_bf16(false, (Adata), false, (Bdata), \
                                            (short)0, (Cacc), false, false)

// ---------------------------------------------------------------------------
// Weight packing: f32 [Nsrc][K] row-major -> bf16 WMMA B-tiles, N padded to
// a multiple of 16 (rows >= Nsrc zero-filled).
// Tile (nt,kt) covers N in [nt*16,nt*16+16), K in [kt*32,kt*32+32).
// Within a tile: lane l holds 16 halves at tile_base + l*16:
//   N = nt*16 + (l&15);  K = kt*32 + (j<8 ? 8h+j : 16+8h+(j-8)), h = l>>4, j = 0..15
// ---------------------------------------------------------------------------
__global__ __launch_bounds__(256) void pack_weights_bf16(
    const float* __restrict__ W, uint16_t* __restrict__ Wp, int K, int Nsrc)
{
    int tile = blockIdx.x;
    int ntk  = K >> 5;
    int nt   = tile / ntk, kt = tile - nt * ntk;
    for (int i = threadIdx.x; i < 512; i += 256) {
        int lane = i >> 4, j = i & 15;
        int n = nt * 16 + (lane & 15);
        int k = kt * 32 + ((j < 8) ? ((lane >> 4) * 8 + j) : (16 + (lane >> 4) * 8 + (j - 8)));
        float f = (n < Nsrc) ? W[n * K + k] : 0.f;
        __bf16 h = (__bf16)f;
        Wp[(size_t)tile * 512 + i] = __builtin_bit_cast(uint16_t, h);
    }
}

// ---------------------------------------------------------------------------
// One 16x16 f32 tile of  X[16xK] @ W^T  via v_wmma_f32_16x16x32_bf16.
// Single live accumulator, kt-inner: this shape compiles spill-free and lets
// the backend software-pipeline ds_load_b128 (A) + global_load_b128 (B)
// against the WMMAs (verified in round-1 disassembly).
// A (16x32 bf16) from LDS, row-major [16][Kdim] halves:
//   lane<16 : row=lane,    K halves {0..7} and {16..23}   -> +0,  +32 bytes
//   lane>=16: row=lane-16, K halves {8..15} and {24..31}  -> +16, +48 bytes
// B: 32 contiguous bytes per lane per packed tile.
// ---------------------------------------------------------------------------
__device__ __forceinline__ v8f tile_mm(v8f c, const __bf16* xl, int Kdim,
                                       const uint16_t* wp, int nt, int ntk,
                                       int lane)
{
    const int  row    = lane & 15;
    const int  hi     = lane >> 4;
    const char* abase = (const char*)xl + ((size_t)(row * Kdim) << 1) + (hi << 4);
    const char* bbase = (const char*)wp + ((size_t)nt * ntk << 10) + (lane << 5);
    for (int kt = 0; kt < ntk; ++kt) {
        ABPack a, b;
        const char* ab = abase + (kt << 6);
        a.q[0] = *(const v4u*)(ab);
        a.q[1] = *(const v4u*)(ab + 32);
        const char* bb = bbase + ((size_t)kt << 10);
        b.q[0] = *(const v4u*)(bb);
        b.q[1] = *(const v4u*)(bb + 16);
        __builtin_prefetch(bb + 1024, 0, 1);   // global_prefetch_b8: next K tile
        c = WMMA_BF16(a.v, b.v, c);
    }
    return c;
}

// ---------------------------------------------------------------------------
// One GRU layer for a 16-row batch tile.
// Gate columns [r|z|n] over 3H=1536 -> 96 N-tiles; wave w owns nt = w, w+8,...
// For r/z tiles gi and gh share one accumulator; for n tiles i_n / h_n are
// kept separate (needed for n = tanh(i_n + r*h_n)).
// Then elementwise: h = (1-z)*tanh(i_n + r*h_n) + z*h.
// ---------------------------------------------------------------------------
__device__ __forceinline__ void gru_layer(
    const __bf16* __restrict__ xl, int Kx,
    const uint16_t* __restrict__ Wihp, const uint16_t* __restrict__ Whhp,
    const float* __restrict__ b_ih, const float* __restrict__ b_hh,
    __bf16* __restrict__ hl,
    __bf16* __restrict__ rL, __bf16* __restrict__ zL,
    float* __restrict__ inL, float* __restrict__ hnL, int tid)
{
    const int wave  = tid >> 5, lane = tid & 31;
    const int ktx   = Kx >> 5;
    const int coll  = lane & 15;
    const int rbase = (lane >> 4) << 3;          // rows 0..7 / 8..15

    for (int nt = wave; nt < NTILES; nt += 8) {
        const int ng = nt * 16 + coll;           // gate column in [0,1536)
        if (nt < 64) {
            // r or z tile: s = gi + gh + biases -> sigmoid, stored bf16.
            v8f c = {};
            c = tile_mm(c, xl, Kx,  Wihp, nt, ktx,      lane);
            c = tile_mm(c, hl, HID, Whhp, nt, HID / 32, lane);
            const float bias = b_ih[ng] + b_hh[ng];
            __bf16* dst = (nt < 32) ? rL : zL;   // wave-uniform select
            const int col  = (nt < 32) ? ng : (ng - HID);
            #pragma unroll
            for (int e = 0; e < 8; ++e) {
                float s = c[e] + bias;
                dst[(rbase + e) * HID + col] = (__bf16)(1.f / (1.f + __expf(-s)));
            }
        } else {
            // n tile: keep i_n and h_n separate.
            v8f ci = {}, ch = {};
            ci = tile_mm(ci, xl, Kx,  Wihp, nt, ktx,      lane);
            ch = tile_mm(ch, hl, HID, Whhp, nt, HID / 32, lane);
            const int col = ng - 2 * HID;
            const float bi = b_ih[ng], bh = b_hh[ng];
            #pragma unroll
            for (int e = 0; e < 8; ++e) {
                inL[(rbase + e) * HID + col] = ci[e] + bi;
                hnL[(rbase + e) * HID + col] = ch[e] + bh;
            }
        }
    }
    __syncthreads();

    // Elementwise combine.
    for (int i = tid; i < ROWS * HID; i += 256) {
        float r = (float)rL[i], z = (float)zL[i];
        float n = tanhf(inL[i] + r * hnL[i]);
        float h = (float)hl[i];
        hl[i] = (__bf16)((1.f - z) * n + z * h);
    }
    __syncthreads();
}

// ---------------------------------------------------------------------------
// Persistent kernel: each block runs 16 batch rows through all T steps.
// ---------------------------------------------------------------------------
__global__ __launch_bounds__(256) void gru_forward(
    const int*   __restrict__ target,
    const float* __restrict__ emb,
    const float* __restrict__ b_ih1, const float* __restrict__ b_hh1,
    const float* __restrict__ b_ih2, const float* __restrict__ b_hh2,
    const float* __restrict__ b_ih3, const float* __restrict__ b_hh3,
    const float* __restrict__ b_out,
    const uint16_t* __restrict__ Wih1p, const uint16_t* __restrict__ Whh1p,
    const uint16_t* __restrict__ Wih2p, const uint16_t* __restrict__ Whh2p,
    const uint16_t* __restrict__ Wih3p, const uint16_t* __restrict__ Whh3p,
    const uint16_t* __restrict__ Woutp,
    float* __restrict__ out)
{
    extern __shared__ char smem[];
    float*  inL     = (float*)smem;                 // 16*512 f32
    float*  hnL     = inL + ROWS * HID;             // 16*512 f32
    float*  logitsL = hnL + ROWS * HID;             // 16*36  f32 (V=33 padded)
    float*  lpL     = logitsL + ROWS * 36;          // 16 f32
    int*    tokL    = (int*)(lpL + ROWS);           // 16 int
    int*    tgtL    = tokL + ROWS;                  // 16 int
    __bf16* rL      = (__bf16*)(tgtL + ROWS);       // 16*512 bf16
    __bf16* zL      = rL  + ROWS * HID;
    __bf16* h1L     = zL  + ROWS * HID;
    __bf16* h2L     = h1L + ROWS * HID;
    __bf16* h3L     = h2L + ROWS * HID;
    __bf16* xeL     = h3L + ROWS * HID;             // 16*256 bf16

    const int tid   = threadIdx.x;
    const int wave  = tid >> 5, lane = tid & 31;
    const int coll  = lane & 15;
    const int rbase = (lane >> 4) << 3;
    const int brow0 = blockIdx.x * ROWS;

    for (int i = tid; i < ROWS * HID; i += 256) {
        h1L[i] = (__bf16)0.f; h2L[i] = (__bf16)0.f; h3L[i] = (__bf16)0.f;
    }
    if (tid < ROWS) lpL[tid] = 0.f;
    __syncthreads();

    for (int t = 0; t < SEQT; ++t) {
        // Teacher-forced input token + target token for this step.
        if (tid < ROWS) {
            int b = brow0 + tid;
            tokL[tid] = (t == 0) ? 0 : target[b * SEQT + t - 1];
            tgtL[tid] = target[b * SEQT + t];
        }
        __syncthreads();
        // Embedding gather -> bf16 A tile [16][256]
        for (int i = tid; i < ROWS * EMB; i += 256) {
            int r = i >> 8, c = i & (EMB - 1);
            xeL[i] = (__bf16)emb[tokL[r] * EMB + c];
        }
        __syncthreads();

        gru_layer(xeL, EMB, Wih1p, Whh1p, b_ih1, b_hh1, h1L, rL, zL, inL, hnL, tid);
        gru_layer(h1L, HID, Wih2p, Whh2p, b_ih2, b_hh2, h2L, rL, zL, inL, hnL, tid);
        gru_layer(h2L, HID, Wih3p, Whh3p, b_ih3, b_hh3, h3L, rL, zL, inL, hnL, tid);

        // Output head via WMMA: logits[16][48] = h3 @ W_out^T, V padded to 48.
        // Waves 0..2 each own one N-tile; only columns < VOCAB are stored.
        if (wave < 3) {
            v8f c = {};
            c = tile_mm(c, h3L, HID, Woutp, wave, HID / 32, lane);
            const int col = wave * 16 + coll;
            if (col < VOCAB) {
                const float bo = b_out[col];
                #pragma unroll
                for (int e = 0; e < 8; ++e)
                    logitsL[(rbase + e) * 36 + col] = c[e] + bo;
            }
        }
        __syncthreads();
        // log_softmax + gather target, accumulate log-prob
        if (tid < ROWS) {
            float m = -1e30f;
            for (int v = 0; v < VOCAB; ++v) m = fmaxf(m, logitsL[tid * 36 + v]);
            float se = 0.f;
            for (int v = 0; v < VOCAB; ++v) se += __expf(logitsL[tid * 36 + v] - m);
            lpL[tid] += logitsL[tid * 36 + tgtL[tid]] - m - __logf(se);
        }
        __syncthreads();
    }
    if (tid < ROWS) out[brow0 + tid] = lpL[tid];
}

// ---------------------------------------------------------------------------
extern "C" void kernel_launch(void* const* d_in, const int* in_sizes, int n_in,
                              void* d_out, int out_size, void* d_ws, size_t ws_size,
                              hipStream_t stream)
{
    const int*   target = (const int*)  d_in[0];
    const float* emb    = (const float*)d_in[1];
    const float* W_ih1  = (const float*)d_in[2];
    const float* W_hh1  = (const float*)d_in[3];
    const float* b_ih1  = (const float*)d_in[4];
    const float* b_hh1  = (const float*)d_in[5];
    const float* W_ih2  = (const float*)d_in[6];
    const float* W_hh2  = (const float*)d_in[7];
    const float* b_ih2  = (const float*)d_in[8];
    const float* b_hh2  = (const float*)d_in[9];
    const float* W_ih3  = (const float*)d_in[10];
    const float* W_hh3  = (const float*)d_in[11];
    const float* b_ih3  = (const float*)d_in[12];
    const float* b_hh3  = (const float*)d_in[13];
    const float* W_out  = (const float*)d_in[14];
    const float* b_out  = (const float*)d_in[15];

    // Workspace: packed bf16 weights (~8.3 MB) — resident in L2 at runtime.
    uint16_t* ws    = (uint16_t*)d_ws;
    uint16_t* Wih1p = ws;
    uint16_t* Whh1p = Wih1p + (size_t)GATES * EMB;
    uint16_t* Wih2p = Whh1p + (size_t)GATES * HID;
    uint16_t* Whh2p = Wih2p + (size_t)GATES * HID;
    uint16_t* Wih3p = Whh2p + (size_t)GATES * HID;
    uint16_t* Whh3p = Wih3p + (size_t)GATES * HID;
    uint16_t* Woutp = Whh3p + (size_t)GATES * HID;  // [VPAD x HID]

    pack_weights_bf16<<<NTILES * (EMB / 32), 256, 0, stream>>>(W_ih1, Wih1p, EMB, GATES);
    pack_weights_bf16<<<NTILES * (HID / 32), 256, 0, stream>>>(W_hh1, Whh1p, HID, GATES);
    pack_weights_bf16<<<NTILES * (HID / 32), 256, 0, stream>>>(W_ih2, Wih2p, HID, GATES);
    pack_weights_bf16<<<NTILES * (HID / 32), 256, 0, stream>>>(W_hh2, Whh2p, HID, GATES);
    pack_weights_bf16<<<NTILES * (HID / 32), 256, 0, stream>>>(W_ih3, Wih3p, HID, GATES);
    pack_weights_bf16<<<NTILES * (HID / 32), 256, 0, stream>>>(W_hh3, Whh3p, HID, GATES);
    pack_weights_bf16<<<(VPAD / 16) * (HID / 32), 256, 0, stream>>>(W_out, Woutp, HID, VOCAB);

    const size_t lds =
        (size_t)ROWS * HID * sizeof(float) * 2      // inL, hnL
      + (size_t)ROWS * 36 * sizeof(float)           // logits (padded)
      + (size_t)ROWS * sizeof(float)                // lp
      + (size_t)ROWS * sizeof(int) * 2              // tok, tgt
      + (size_t)ROWS * HID * 2 * 5                  // rL, zL, h1, h2, h3 (bf16)
      + (size_t)ROWS * EMB * 2;                     // xeL (bf16)  => 158,144 B

    (void)hipFuncSetAttribute(reinterpret_cast<const void*>(gru_forward),
                              hipFuncAttributeMaxDynamicSharedMemorySize,
                              (int)lds);

    gru_forward<<<BATCH / ROWS, 256, lds, stream>>>(
        target, emb,
        b_ih1, b_hh1, b_ih2, b_hh2, b_ih3, b_hh3,
        b_out,
        Wih1p, Whh1p, Wih2p, Whh2p, Wih3p, Whh3p, Woutp,
        (float*)d_out);
}
